// StateModelEncoder_11922829214272
// MI455X (gfx1250) — matrix operand order repaired and checked by reading, not compiled, not run on hardware
//
#include <hip/hip_runtime.h>
#include <hip/hip_bf16.h>

// ---------------------------------------------------------------------------
// CDNA5 (gfx1250) GNN forward pass.
//  - 64x64 GEMMs: v_wmma_f32_16x16x32_f16, W staged into LDS via the Tensor
//    Data Mover (tensor_load_to_lds + s_wait_tensorcnt), B fragments from LDS.
//  - Edge message passing: coalesced gathers + global_atomic_add_f32 scatters.
// ---------------------------------------------------------------------------

typedef __attribute__((ext_vector_type(16))) _Float16 v16h;
typedef __attribute__((ext_vector_type(8)))  float    v8f;
typedef unsigned int v4u  __attribute__((ext_vector_type(4)));
typedef int          v8i_ __attribute__((ext_vector_type(8)));
typedef int          v4i_ __attribute__((ext_vector_type(4)));

#define HID 64

// ---------------- WMMA GEMM: out[N,64] (+)= x[N,64] @ W[64,64] (+bias)(+relu)
// One wave per 16x16 output tile; K=64 -> two v_wmma_f32_16x16x32_f16.
// W (16 KB f32) is DMA'd into LDS once per block by the TDM.
__global__ __launch_bounds__(256)
void gemm64_wmma_kernel(const float* __restrict__ x, const float* __restrict__ W,
                        const float* __restrict__ bias, float* __restrict__ out,
                        int ntiles, int beta, int do_relu)
{
    __shared__ float sW[HID * HID];

    int lane = threadIdx.x & 31;
    int wave = threadIdx.x >> 5;
    int tile = blockIdx.x * 8 + wave;
    bool active = tile < ntiles;

    // ---- TDM: load W[64,64] f32 (16 KB) into LDS, wave 0 issues descriptor.
    if (wave == 0) {
        unsigned long long ga = (unsigned long long)(uintptr_t)W;
        unsigned int lds = (unsigned int)(uintptr_t)&sW[0];
        v4u g0;
        g0[0] = 1u;                                   // count=1, no gather
        g0[1] = lds;                                  // lds_addr (bytes)
        g0[2] = (unsigned int)ga;                     // global_addr[31:0]
        g0[3] = (unsigned int)((ga >> 32) & 0x1FFFFFFu) | (2u << 30); // addr hi | type=2
        v8i_ g1;
        g1[0] = 0x20000;                              // data_size=4B, mask=0
        g1[1] = 64 << 16;                             // tensor_dim0 = 64 (lo16)
        g1[2] = 64 << 16;                             // tensor_dim1 = 64 (lo16)
        g1[3] = 64 << 16;                             // tile_dim0 = 64
        g1[4] = 64;                                   // tile_dim1 = 64, tile_dim2 = 0
        g1[5] = 64;                                   // tensor_dim0_stride = 64
        g1[6] = (int)(4096u << 16);                   // tensor_dim1_stride = 4096 (lo16)
        g1[7] = 0;
        v4i_ gz = {0, 0, 0, 0};
#if __clang_major__ >= 23
        v8i_ gz8 = {0, 0, 0, 0, 0, 0, 0, 0};
        __builtin_amdgcn_tensor_load_to_lds(g0, g1, gz, gz, gz8, 0);
#else
        __builtin_amdgcn_tensor_load_to_lds(g0, g1, gz, gz, 0);
#endif
        __builtin_amdgcn_s_wait_tensorcnt(0);
    }
    __syncthreads();
    if (!active) return;                   // wave-uniform: EXEC all-1 below

    int row0 = (tile >> 2) << 4;           // 4 column tiles per 16-row block
    int col0 = (tile & 3) << 4;
    int hl   = lane >> 4;                  // lane half (0/1)
    int l16  = lane & 15;

    const float* xrow = x + (size_t)(row0 + l16) * HID;   // A row M = l16

    v8f c = {};
    if (beta) {
#pragma unroll
        for (int v = 0; v < 8; ++v)
            c[v] = out[(size_t)(row0 + v + 8 * hl) * HID + col0 + l16];
    }

#pragma unroll
    for (int kb = 0; kb < 2; ++kb) {
        int kbase = kb * 32;
        v16h a, b;
        // A 16x32 f16 layout: element t -> K = kbase + 16*hl + t, M = l16
#pragma unroll
        for (int j = 0; j < 16; ++j)
            a[j] = (_Float16)xrow[kbase + 16 * hl + j];
        // B 32x16 f16 layout: element t -> K = kbase + 16*hl + t, N = l16
#pragma unroll
        for (int t = 0; t < 16; ++t)
            b[t] = (_Float16)sW[(kbase + 16 * hl + t) * HID + col0 + l16];

        c = __builtin_amdgcn_wmma_f32_16x16x32_f16(
                false, a, false, b, (short)0, c, false, false);
    }

    if (bias) {
        float bb = bias[col0 + l16];
#pragma unroll
        for (int v = 0; v < 8; ++v) c[v] += bb;
    }
#pragma unroll
    for (int v = 0; v < 8; ++v) {
        float val = c[v];
        if (do_relu) val = fmaxf(val, 0.f);
        out[(size_t)(row0 + v + 8 * hl) * HID + col0 + l16] = val;
    }
}

// ---------------- small-K dense: out[N,64] (+)= x[N,K] @ W[K,64] (+bias)(+relu)
__global__ void small_gemm_kernel(const float* __restrict__ x, const float* __restrict__ W,
                                  const float* __restrict__ bias, float* __restrict__ out,
                                  int nrows, int K, int beta, int do_relu)
{
    int tid = blockIdx.x * blockDim.x + threadIdx.x;
    if (tid >= nrows * HID) return;
    int row = tid >> 6, col = tid & 63;
    float acc = beta ? out[tid] : 0.f;
    for (int k = 0; k < K; ++k) acc += x[row * K + k] * W[k * HID + col];
    if (bias) acc += bias[col];
    if (do_relu) acc = fmaxf(acc, 0.f);
    out[tid] = acc;
}

// ---------------- graph helpers --------------------------------------------
__global__ void deg_count_kernel(const int* __restrict__ dst, int E, float* deg)
{
    int e = blockIdx.x * blockDim.x + threadIdx.x;
    if (e < E) atomicAdd(&deg[dst[e]], 1.0f);
}

__global__ void dinv_kernel(float* deg, int n)
{
    int i = blockIdx.x * blockDim.x + threadIdx.x;
    if (i < n) { float d = deg[i]; deg[i] = d > 0.f ? rsqrtf(d) : 0.f; }
}

__global__ void gcnw_kernel(const int* __restrict__ src, const int* __restrict__ dst,
                            const float* __restrict__ dinv, float* w, int E)
{
    int e = blockIdx.x * blockDim.x + threadIdx.x;
    if (e < E) w[e] = dinv[src[e]] * dinv[dst[e]];
}

// hout[dst,f] += w[e] * hin[src,f]  (F-wide features)
__global__ void weighted_scatter_kernel(const int* __restrict__ src, const int* __restrict__ dst,
                                        const float* __restrict__ w,
                                        const float* __restrict__ hin, float* hout,
                                        int E, int F)
{
    int tid = blockIdx.x * blockDim.x + threadIdx.x;
    if (tid >= E * F) return;
    int e = tid / F, f = tid - e * F;
    atomicAdd(&hout[(size_t)dst[e] * F + f], w[e] * hin[(size_t)src[e] * F + f]);
}

// per-relation RGCN accumulate + count
__global__ void rgcn_acc_kernel(const int* __restrict__ src, const int* __restrict__ dst,
                                const int* __restrict__ et, int r,
                                const float* __restrict__ g, float* acc, float* cnt, int E)
{
    int tid = blockIdx.x * blockDim.x + threadIdx.x;
    if (tid >= E * HID) return;
    int e = tid >> 6, f = tid & 63;
    if (et[e] != r) return;
    atomicAdd(&acc[(size_t)dst[e] * HID + f], g[(size_t)src[e] * HID + f]);
    if (f == 0) atomicAdd(&cnt[dst[e]], 1.0f);
}

// SAGE mean aggregation numerator + count
__global__ void mean_agg_kernel(const int* __restrict__ src, const int* __restrict__ dst,
                                const float* __restrict__ xin, float* acc, float* cnt, int E)
{
    int tid = blockIdx.x * blockDim.x + threadIdx.x;
    if (tid >= E * HID) return;
    int e = tid >> 6, f = tid & 63;
    atomicAdd(&acc[(size_t)dst[e] * HID + f], xin[(size_t)src[e] * HID + f]);
    if (f == 0) atomicAdd(&cnt[dst[e]], 1.0f);
}

__global__ void rowdiv_kernel(float* x, const float* __restrict__ cnt, int nrows)
{
    int tid = blockIdx.x * blockDim.x + threadIdx.x;
    if (tid >= nrows * HID) return;
    x[tid] /= fmaxf(cnt[tid >> 6], 1.f);
}

// ResGated edge update: out[dst,f] += sigmoid(k[dst,f]+q[src,f]+e[f]) * v[src,f]
__global__ void resgated_kernel(const int* __restrict__ src, const int* __restrict__ dst,
                                const float* __restrict__ ea,
                                const float* __restrict__ kk, const float* __restrict__ q,
                                const float* __restrict__ vv,
                                const float* __restrict__ We, const float* __restrict__ be,
                                float* out, int E)
{
    int tid = blockIdx.x * blockDim.x + threadIdx.x;
    if (tid >= E * HID) return;
    int e = tid >> 6, f = tid & 63;
    int s = src[e], d = dst[e];
    float et = ea[e * 2 + 0] * We[f] + ea[e * 2 + 1] * We[HID + f] + be[f];
    float z  = kk[(size_t)d * HID + f] + q[(size_t)s * HID + f] + et;
    float gate = 1.f / (1.f + expf(-z));
    atomicAdd(&out[(size_t)d * HID + f], gate * vv[(size_t)s * HID + f]);
}

__global__ void relu_kernel(float* x, int n)
{
    int i = blockIdx.x * blockDim.x + threadIdx.x;
    if (i < n) x[i] = fmaxf(x[i], 0.f);
}

__global__ void l2norm_relu_kernel(float* x, int nrows)
{
    int row = blockIdx.x * blockDim.x + threadIdx.x;
    if (row >= nrows) return;
    float ss = 0.f;
    for (int f = 0; f < HID; ++f) { float v = x[(size_t)row * HID + f]; ss += v * v; }
    float inv = 1.f / fmaxf(sqrtf(ss), 1e-12f);
    for (int f = 0; f < HID; ++f)
        x[(size_t)row * HID + f] = fmaxf(x[(size_t)row * HID + f] * inv, 0.f);
}

// relu(s @ lin_W[64,8] + lin_b)
__global__ void lin8_kernel(const float* __restrict__ s, const float* __restrict__ W,
                            const float* __restrict__ b, float* out, int nrows)
{
    int tid = blockIdx.x * blockDim.x + threadIdx.x;
    if (tid >= nrows * 8) return;
    int row = tid >> 3, col = tid & 7;
    float acc = b[col];
    for (int k = 0; k < HID; ++k) acc += s[(size_t)row * HID + k] * W[k * 8 + col];
    out[tid] = fmaxf(acc, 0.f);
}

__global__ void logits_kernel(const float* __restrict__ f8, const float* __restrict__ lw,
                              const float* __restrict__ lb, float* logits, int n)
{
    int i = blockIdx.x * blockDim.x + threadIdx.x;
    if (i >= n) return;
    float a = lb[0];
    for (int k = 0; k < 8; ++k) a += f8[i * 8 + k] * lw[k];
    logits[i] = a;
}

__global__ void reduce_max_kernel(const float* __restrict__ x, int n, float* res)
{
    __shared__ float sm[256];
    float m = -3.4e38f;
    for (int i = threadIdx.x; i < n; i += 256) m = fmaxf(m, x[i]);
    sm[threadIdx.x] = m; __syncthreads();
    for (int s = 128; s > 0; s >>= 1) {
        if (threadIdx.x < s) sm[threadIdx.x] = fmaxf(sm[threadIdx.x], sm[threadIdx.x + s]);
        __syncthreads();
    }
    if (threadIdx.x == 0) res[0] = sm[0];
}

__global__ void reduce_lse_kernel(const float* __restrict__ x, int n,
                                  const float* __restrict__ gmax, float* res)
{
    __shared__ float sm[256];
    float m = gmax[0], s = 0.f;
    for (int i = threadIdx.x; i < n; i += 256) s += expf(x[i] - m);
    sm[threadIdx.x] = s; __syncthreads();
    for (int t = 128; t > 0; t >>= 1) {
        if (threadIdx.x < t) sm[threadIdx.x] += sm[threadIdx.x + t];
        __syncthreads();
    }
    if (threadIdx.x == 0) res[0] = logf(sm[0]) + m;  // logsumexp
}

__global__ void logsoftmax_kernel(const float* __restrict__ logits,
                                  const float* __restrict__ lse, float* out, int n)
{
    int i = blockIdx.x * blockDim.x + threadIdx.x;
    if (i < n) out[i] = logits[i] - lse[0];
}

// ---------------------------------------------------------------------------
static inline int gs(long long n, int b) { return (int)((n + b - 1) / b); }

extern "C" void kernel_launch(void* const* d_in, const int* in_sizes, int n_in,
                              void* d_out, int out_size, void* d_ws, size_t ws_size,
                              hipStream_t stream)
{
    // ---- inputs (setup_inputs dict order, params expanded in insertion order)
    const float* game_x  = (const float*)d_in[0];
    const float* state_x = (const float*)d_in[1];
    const int*   evv     = (const int*)d_in[2];
    const int*   et_vv   = (const int*)d_in[3];
    const int*   eh      = (const int*)d_in[4];
    const float* ea_h    = (const float*)d_in[5];
    const int*   ein     = (const int*)d_in[6];
    const int*   ess     = (const int*)d_in[7];

    const float* tag1_W   = (const float*)d_in[8];
    const float* tag1_b   = (const float*)d_in[9];
    const float* rgcn_W   = (const float*)d_in[10];
    const float* rgcn_root= (const float*)d_in[11];
    const float* rgcn_b   = (const float*)d_in[12];
    const float* rg_Wk    = (const float*)d_in[13];
    const float* rg_bk    = (const float*)d_in[14];
    const float* rg_Wq    = (const float*)d_in[15];
    const float* rg_bq    = (const float*)d_in[16];
    const float* rg_Wv    = (const float*)d_in[17];
    const float* rg_bv    = (const float*)d_in[18];
    const float* rg_We    = (const float*)d_in[19];
    const float* rg_be    = (const float*)d_in[20];
    const float* rg_Wskip = (const float*)d_in[21];
    const float* rg_bias  = (const float*)d_in[22];
    const float* s32_Wl   = (const float*)d_in[23];
    const float* s32_bl   = (const float*)d_in[24];
    const float* s32_Wr   = (const float*)d_in[25];
    const float* s4_Wl    = (const float*)d_in[26];
    const float* s4_bl    = (const float*)d_in[27];
    const float* s4_Wr    = (const float*)d_in[28];
    const float* s42_Wl   = (const float*)d_in[29];
    const float* s42_bl   = (const float*)d_in[30];
    const float* s42_Wr   = (const float*)d_in[31];
    const float* tag2_W   = (const float*)d_in[32];
    const float* tag2_b   = (const float*)d_in[33];
    const float* s5_Wl    = (const float*)d_in[34];
    const float* s5_bl    = (const float*)d_in[35];
    const float* s5_Wr    = (const float*)d_in[36];
    const float* lin_W    = (const float*)d_in[37];
    const float* lin_b    = (const float*)d_in[38];
    const float* last_W   = (const float*)d_in[39];
    const float* last_b   = (const float*)d_in[40];

    const int NV  = in_sizes[0] / 7;
    const int NS  = in_sizes[1] / 7;
    const int EVV = in_sizes[2] / 2;
    const int EH  = in_sizes[4] / 2;
    const int EIN = in_sizes[6] / 2;
    const int ESS = in_sizes[7] / 2;

    const int *svv = evv, *dvv = evv + EVV;
    const int *sh  = eh,  *dh  = eh  + EH;
    const int *sin_ = ein, *din_ = ein + EIN;
    const int *sss = ess, *dss = ess + ESS;

    // ---- workspace carve-out (floats)
    float* ws = (float*)d_ws;
    size_t off = 0;
    auto alloc = [&](size_t n) { float* p = ws + off; off += n; return p; };
    float* bufA  = alloc((size_t)NV * HID);
    float* bufB  = alloc((size_t)NV * HID);
    float* bufC  = alloc((size_t)NV * HID);
    float* h7a   = alloc((size_t)NV * 7);
    float* h7b   = alloc((size_t)NV * 7);
    float* wvv   = alloc(EVV);
    float* degv  = alloc(NV);
    float* cnt   = alloc(NV);              // reused for NS-sized counts
    float* sA    = alloc((size_t)NS * HID);
    float* sB    = alloc((size_t)NS * HID);
    float* sC    = alloc((size_t)NS * HID);
    float* sD    = alloc((size_t)NS * HID);
    float* feat8 = alloc((size_t)NS * 8);
    float* logit = alloc(NS);
    float* wss   = alloc(ESS);
    float* scal  = alloc(4);

    const int B = 256;
    const int tilesV = (NV / 16) * 4;      // NV, NS are multiples of 16
    const int tilesS = (NS / 16) * 4;

    auto gemm64 = [&](const float* x, const float* W, const float* bias,
                      float* out, int ntiles, int beta, int relu) {
        gemm64_wmma_kernel<<<gs(ntiles, 8), 256, 0, stream>>>(x, W, bias, out, ntiles, beta, relu);
    };

    // ================= Stage 1: tag1 (game_x [NV,7] -> g in bufA) ===========
    hipMemsetAsync(degv, 0, (size_t)NV * 4, stream);
    deg_count_kernel<<<gs(EVV, B), B, 0, stream>>>(dvv, EVV, degv);
    dinv_kernel<<<gs(NV, B), B, 0, stream>>>(degv, NV);
    gcnw_kernel<<<gs(EVV, B), B, 0, stream>>>(svv, dvv, degv, wvv, EVV);

    small_gemm_kernel<<<gs((long long)NV * HID, B), B, 0, stream>>>(game_x, tag1_W, nullptr, bufA, NV, 7, 0, 0);
    hipMemsetAsync(h7a, 0, (size_t)NV * 7 * 4, stream);
    weighted_scatter_kernel<<<gs((long long)EVV * 7, B), B, 0, stream>>>(svv, dvv, wvv, game_x, h7a, EVV, 7);
    small_gemm_kernel<<<gs((long long)NV * HID, B), B, 0, stream>>>(h7a, tag1_W + 7 * HID, nullptr, bufA, NV, 7, 1, 0);
    hipMemsetAsync(h7b, 0, (size_t)NV * 7 * 4, stream);
    weighted_scatter_kernel<<<gs((long long)EVV * 7, B), B, 0, stream>>>(svv, dvv, wvv, h7a, h7b, EVV, 7);
    small_gemm_kernel<<<gs((long long)NV * HID, B), B, 0, stream>>>(h7b, tag1_W + 14 * HID, tag1_b, bufA, NV, 7, 1, 1);
    // g = bufA

    // ================= Stage 2: RGCN (bufA -> bufB) =========================
    gemm64(bufA, rgcn_root, rgcn_b, bufB, tilesV, 0, 0);
    for (int r = 0; r < 3; ++r) {
        hipMemsetAsync(bufC, 0, (size_t)NV * HID * 4, stream);
        hipMemsetAsync(cnt, 0, (size_t)NV * 4, stream);
        rgcn_acc_kernel<<<gs((long long)EVV * HID, B), B, 0, stream>>>(svv, dvv, et_vv, r, bufA, bufC, cnt, EVV);
        rowdiv_kernel<<<gs((long long)NV * HID, B), B, 0, stream>>>(bufC, cnt, NV);
        gemm64(bufC, rgcn_W + (size_t)r * HID * HID, nullptr, bufB, tilesV, 1, r == 2);
    }
    // g = bufB

    // ================= Stage 3: ResGated (g, state_x -> s in sA) ============
    small_gemm_kernel<<<gs((long long)NS * HID, B), B, 0, stream>>>(state_x, rg_Wk, rg_bk, sB, NS, 7, 0, 0);   // k
    gemm64(bufB, rg_Wq, rg_bq, bufA, tilesV, 0, 0);                                                            // q
    gemm64(bufB, rg_Wv, rg_bv, bufC, tilesV, 0, 0);                                                            // v
    small_gemm_kernel<<<gs((long long)NS * HID, B), B, 0, stream>>>(state_x, rg_Wskip, rg_bias, sA, NS, 7, 0, 0);
    resgated_kernel<<<gs((long long)EH * HID, B), B, 0, stream>>>(sh, dh, ea_h, sB, bufA, bufC, rg_We, rg_be, sA, EH);
    relu_kernel<<<gs((long long)NS * HID, B), B, 0, stream>>>(sA, NS * HID);
    // s = sA

    // ================= Stage 4: SAGE s32 over e_h ===========================
    hipMemsetAsync(sB, 0, (size_t)NS * HID * 4, stream);
    hipMemsetAsync(cnt, 0, (size_t)NS * 4, stream);
    mean_agg_kernel<<<gs((long long)EH * HID, B), B, 0, stream>>>(sh, dh, bufB, sB, cnt, EH);
    rowdiv_kernel<<<gs((long long)NS * HID, B), B, 0, stream>>>(sB, cnt, NS);
    gemm64(sB, s32_Wl, s32_bl, sC, tilesS, 0, 0);
    gemm64(sA, s32_Wr, nullptr, sC, tilesS, 1, 0);
    l2norm_relu_kernel<<<gs(NS, B), B, 0, stream>>>(sC, NS);
    // s = sC

    // ================= Stage 5: SAGE s4 over e_in ===========================
    hipMemsetAsync(sB, 0, (size_t)NS * HID * 4, stream);
    hipMemsetAsync(cnt, 0, (size_t)NS * 4, stream);
    mean_agg_kernel<<<gs((long long)EIN * HID, B), B, 0, stream>>>(sin_, din_, bufB, sB, cnt, EIN);
    rowdiv_kernel<<<gs((long long)NS * HID, B), B, 0, stream>>>(sB, cnt, NS);
    gemm64(sB, s4_Wl, s4_bl, sA, tilesS, 0, 0);
    gemm64(sC, s4_Wr, nullptr, sA, tilesS, 1, 0);
    l2norm_relu_kernel<<<gs(NS, B), B, 0, stream>>>(sA, NS);
    // s = sA

    // ================= Stage 6: SAGE s42 over e_in (same g-aggregation sB) ==
    gemm64(sB, s42_Wl, s42_bl, sC, tilesS, 0, 0);
    gemm64(sA, s42_Wr, nullptr, sC, tilesS, 1, 0);
    l2norm_relu_kernel<<<gs(NS, B), B, 0, stream>>>(sC, NS);
    // s = sC

    // ================= Stage 7: tag2 over e_ss ==============================
    hipMemsetAsync(cnt, 0, (size_t)NS * 4, stream);
    deg_count_kernel<<<gs(ESS, B), B, 0, stream>>>(dss, ESS, cnt);
    dinv_kernel<<<gs(NS, B), B, 0, stream>>>(cnt, NS);
    gcnw_kernel<<<gs(ESS, B), B, 0, stream>>>(sss, dss, cnt, wss, ESS);

    gemm64(sC, tag2_W, nullptr, sA, tilesS, 0, 0);
    hipMemsetAsync(sB, 0, (size_t)NS * HID * 4, stream);
    weighted_scatter_kernel<<<gs((long long)ESS * HID, B), B, 0, stream>>>(sss, dss, wss, sC, sB, ESS, HID);
    gemm64(sB, tag2_W + HID * HID, nullptr, sA, tilesS, 1, 0);
    hipMemsetAsync(sD, 0, (size_t)NS * HID * 4, stream);
    weighted_scatter_kernel<<<gs((long long)ESS * HID, B), B, 0, stream>>>(sss, dss, wss, sB, sD, ESS, HID);
    gemm64(sD, tag2_W + 2 * HID * HID, tag2_b, sA, tilesS, 1, 1);
    // s = sA

    // ================= Stage 8: SAGE s5 over e_ss ===========================
    hipMemsetAsync(sB, 0, (size_t)NS * HID * 4, stream);
    hipMemsetAsync(cnt, 0, (size_t)NS * 4, stream);
    mean_agg_kernel<<<gs((long long)ESS * HID, B), B, 0, stream>>>(sss, dss, sA, sB, cnt, ESS);
    rowdiv_kernel<<<gs((long long)NS * HID, B), B, 0, stream>>>(sB, cnt, NS);
    gemm64(sB, s5_Wl, s5_bl, sC, tilesS, 0, 0);
    gemm64(sA, s5_Wr, nullptr, sC, tilesS, 1, 0);
    l2norm_relu_kernel<<<gs(NS, B), B, 0, stream>>>(sC, NS);
    // s = sC

    // ================= Stage 9/10: heads + global log-softmax ===============
    lin8_kernel<<<gs((long long)NS * 8, B), B, 0, stream>>>(sC, lin_W, lin_b, feat8, NS);
    logits_kernel<<<gs(NS, B), B, 0, stream>>>(feat8, last_W, last_b, logit, NS);
    reduce_max_kernel<<<1, 256, 0, stream>>>(logit, NS, scal + 1);
    reduce_lse_kernel<<<1, 256, 0, stream>>>(logit, NS, scal + 1, scal);
    logsoftmax_kernel<<<gs(NS, B), B, 0, stream>>>(logit, scal, (float*)d_out, NS);

    (void)n_in; (void)out_size; (void)ws_size;
}
